// MultiGraphBlock_69655779607243
// MI455X (gfx1250) — matrix operand order repaired and checked by reading, not compiled, not run on hardware
//
#include <hip/hip_runtime.h>

// ---------------------------------------------------------------------------
// Types for CDNA5 WMMA (wave32)
// ---------------------------------------------------------------------------
typedef __attribute__((ext_vector_type(16))) __bf16 bf16x16;
typedef __attribute__((ext_vector_type(8)))  float  f32x8;
typedef __attribute__((ext_vector_type(4)))  int    i32x4;

union ABFrag { i32x4 q[2]; bf16x16 v; };   // 32 bytes = 8 VGPRs

__device__ __forceinline__ unsigned short f2bf(float f) {
  unsigned u = __builtin_bit_cast(unsigned, f);
  u += 0x7FFFu + ((u >> 16) & 1u);          // round-to-nearest-even
  return (unsigned short)(u >> 16);
}

__device__ __forceinline__ unsigned pack2bf(float a, float b) {
  return (unsigned)f2bf(a) | ((unsigned)f2bf(b) << 16);
}

// ---------------------------------------------------------------------------
// Per-wave GEMM: 16 rows (rt) x 64 cols (col_base..+63) of the output,
// A (bf16) staged in LDS row-major with padded stride `lda`,
// B read directly from global transposed weights Wt[n][k] (bf16, L2 resident).
// A fragment (16-bit A 16x32 layout): two ds_load_b128 per lane.
// B fragment (32x16): two global_load_b128 per lane (k = 8h+s, 16+8h+s).
// ---------------------------------------------------------------------------
template <int KDIM>
__device__ __forceinline__ void wave_gemm(
    const unsigned short* __restrict__ wt,   // [128][KDIM] bf16
    const unsigned short* lA,                // LDS A base [32][lda]
    int lda, int rt, int col_base, int lane, f32x8 acc[4])
{
  const int msel = lane >> 4;      // 0 or 1
  const int l15  = lane & 15;
  const unsigned short* arow = lA + (rt * 16 + l15) * lda;
  for (int koff = 0; koff < KDIM; koff += 32) {
    ABFrag a;
    a.q[0] = *(const i32x4*)(arow + koff + msel * 8);        // ds_load_b128
    a.q[1] = *(const i32x4*)(arow + koff + 16 + msel * 8);   // ds_load_b128
#pragma unroll
    for (int ct = 0; ct < 4; ++ct) {
      ABFrag b;
      const unsigned short* bp =
          wt + (size_t)(col_base + ct * 16 + l15) * KDIM + koff + msel * 8;
      b.q[0] = *(const i32x4*)(bp);         // global_load_b128
      b.q[1] = *(const i32x4*)(bp + 16);    // global_load_b128
      acc[ct] = __builtin_amdgcn_wmma_f32_16x16x32_bf16(
          false, a.v, false, b.v, (short)0, acc[ct], false, false);
    }
  }
}

// C/D layout: VGPR v: lanes 0-15 -> M=v, N=lane; lanes 16-31 -> M=v+8, N=lane-16.
__device__ __forceinline__ void store_relu_bf16(
    unsigned short* lOut, int ldo, const float* __restrict__ bias,
    int rt, int col_base, int lane, const f32x8 acc[4])
{
  const int msel = lane >> 4;
  const int l15  = lane & 15;
#pragma unroll
  for (int ct = 0; ct < 4; ++ct) {
    int n = col_base + ct * 16 + l15;
    float bv = bias[n];
#pragma unroll
    for (int v = 0; v < 8; ++v) {
      int row = rt * 16 + msel * 8 + v;
      float h = acc[ct][v] + bv;
      h = h > 0.f ? h : 0.f;
      lOut[row * ldo + n] = f2bf(h);
    }
  }
}

__device__ __forceinline__ void store_bias_f32(
    float* lOut, int ldo, const float* __restrict__ bias,
    int rt, int col_base, int lane, const f32x8 acc[4])
{
  const int msel = lane >> 4;
  const int l15  = lane & 15;
#pragma unroll
  for (int ct = 0; ct < 4; ++ct) {
    int n = col_base + ct * 16 + l15;
    float bv = bias[n];
#pragma unroll
    for (int v = 0; v < 8; ++v) {
      int row = rt * 16 + msel * 8 + v;
      lOut[row * ldo + n] = acc[ct][v] + bv;
    }
  }
}

// ---------------------------------------------------------------------------
// Edge kernel: 32 edges per block, 128 threads (4 wave32).
// ---------------------------------------------------------------------------
#define BLK 128
#define MT  32
#define KS1E 392   // 384 + 8 pad (bank-conflict avoidance; stride%8==0 for b128)
#define KS2  136   // 128 + 8 pad
#define FS   132   // 128 + 4 pad (f32)

__global__ __launch_bounds__(BLK) void edge_kernel(
    const float* __restrict__ x, const int* __restrict__ eidx,
    float* __restrict__ ea,
    const unsigned short* __restrict__ wt1, const unsigned short* __restrict__ wt2,
    const unsigned short* __restrict__ wt3,
    const float* __restrict__ b1, const float* __restrict__ b2,
    const float* __restrict__ b3,
    const float* __restrict__ gw, const float* __restrict__ bt,
    float* __restrict__ agg, int Ecnt)
{
  __shared__ unsigned short bufA[MT * KS1E];   // 25088 B
  __shared__ unsigned short bufB[MT * KS2];    //  8704 B
  __shared__ float          bufF[MT * FS];     // 16896 B
  __shared__ int sidx[MT], didx[MT];

  const int tid  = threadIdx.x;
  const int lane = tid & 31;
  const int wid  = tid >> 5;           // 0..3
  const int rt   = wid >> 1;           // row tile 0..1
  const int colb = (wid & 1) * 64;     // col half
  const long e0  = (long)blockIdx.x * MT;

  if (tid < MT) {
    long e = e0 + tid;
    sidx[tid] = (e < Ecnt) ? eidx[e] : 0;
    didx[tid] = (e < Ecnt) ? eidx[(long)Ecnt + e] : 0;
  }
  __syncthreads();

  // Stage e_in = cat(x[src], x[dst], ea) as bf16 into LDS: 32 x 384,
  // vectorized: float4 global loads -> packed bf16x4 -> ds_store_b64.
  for (int i = 0; i < (MT * 96) / BLK; ++i) {    // 96 float4 per row; 24 iters
    int f  = tid + BLK * i;
    int r  = f / 96;
    int c4 = f - r * 96;
    int c  = c4 * 4;
    long e = e0 + r;
    float4 v;
    if (c < 128)       v = *(const float4*)(x + (long)sidx[r] * 128 + c);
    else if (c < 256)  v = *(const float4*)(x + (long)didx[r] * 128 + (c - 128));
    else if (e < Ecnt) v = *(const float4*)(ea + e * 128 + (c - 256));
    else               v = make_float4(0.f, 0.f, 0.f, 0.f);
    unsigned* dst = (unsigned*)(bufA + r * KS1E + c);
    dst[0] = pack2bf(v.x, v.y);
    dst[1] = pack2bf(v.z, v.w);
  }
  __syncthreads();

  const f32x8 zero = {0.f,0.f,0.f,0.f,0.f,0.f,0.f,0.f};
  f32x8 acc[4];

  // Layer 1: [32,384] x [384,128]
  acc[0]=zero; acc[1]=zero; acc[2]=zero; acc[3]=zero;
  wave_gemm<384>(wt1, bufA, KS1E, rt, colb, lane, acc);
  store_relu_bf16(bufB, KS2, b1, rt, colb, lane, acc);
  __syncthreads();

  // Layer 2: [32,128] x [128,128] (output into bufA region; safe after barrier)
  acc[0]=zero; acc[1]=zero; acc[2]=zero; acc[3]=zero;
  wave_gemm<128>(wt2, bufB, KS2, rt, colb, lane, acc);
  store_relu_bf16(bufA, KS2, b2, rt, colb, lane, acc);
  __syncthreads();

  // Layer 3: [32,128] x [128,128] -> f32 pre-LN buffer
  acc[0]=zero; acc[1]=zero; acc[2]=zero; acc[3]=zero;
  wave_gemm<128>(wt3, bufA, KS2, rt, colb, lane, acc);
  store_bias_f32(bufF, FS, b3, rt, colb, lane, acc);
  __syncthreads();

  // LayerNorm + residual + in-place ea update + scatter-add into agg.
  // 4 lanes per row, 32 cols each (8 float4); 4-lane shuffle reduction.
  const int r = tid >> 2;
  const int q = tid & 3;
  long e = e0 + r;
  const float4* rowF4 = (const float4*)(bufF + r * FS + q * 32);
  float s = 0.f, ss = 0.f;
#pragma unroll
  for (int i = 0; i < 8; ++i) {
    float4 h = rowF4[i];
    s  += h.x + h.y + h.z + h.w;
    ss += h.x * h.x + h.y * h.y + h.z * h.z + h.w * h.w;
  }
  s  += __shfl_xor(s, 1, 4);   s  += __shfl_xor(s, 2, 4);
  ss += __shfl_xor(ss, 1, 4);  ss += __shfl_xor(ss, 2, 4);
  float mu  = s * (1.f / 128.f);
  float var = ss * (1.f / 128.f) - mu * mu;
  float rs  = rsqrtf(var + 1e-5f);
  if (e < Ecnt) {
    float4*       eaRow4 = (float4*)(ea + e * 128 + q * 32);
    const float4* g4     = (const float4*)(gw + q * 32);
    const float4* bt4    = (const float4*)(bt + q * 32);
    float*        aggRow = agg + (long)didx[r] * 128 + q * 32;
#pragma unroll
    for (int i = 0; i < 8; ++i) {
      float4 h  = rowF4[i];
      float4 gv = g4[i];
      float4 bv = bt4[i];
      float4 ev = eaRow4[i];
      float4 o;
      o.x = ev.x + (h.x - mu) * rs * gv.x + bv.x;
      o.y = ev.y + (h.y - mu) * rs * gv.y + bv.y;
      o.z = ev.z + (h.z - mu) * rs * gv.z + bv.z;
      o.w = ev.w + (h.w - mu) * rs * gv.w + bv.w;
      eaRow4[i] = o;
      atomicAdd(aggRow + i * 4 + 0, o.x);   // segment_sum over dst
      atomicAdd(aggRow + i * 4 + 1, o.y);
      atomicAdd(aggRow + i * 4 + 2, o.z);
      atomicAdd(aggRow + i * 4 + 3, o.w);
    }
  }
}

// ---------------------------------------------------------------------------
// Node kernel: 32 nodes per block, cat(x, agg) -> MLP -> LN -> residual.
// ---------------------------------------------------------------------------
#define KS1N 264   // 256 + 8 pad

__global__ __launch_bounds__(BLK) void node_kernel(
    float* __restrict__ x, const float* __restrict__ agg,
    const unsigned short* __restrict__ wt1, const unsigned short* __restrict__ wt2,
    const unsigned short* __restrict__ wt3,
    const float* __restrict__ b1, const float* __restrict__ b2,
    const float* __restrict__ b3,
    const float* __restrict__ gw, const float* __restrict__ bt,
    int Ncnt)
{
  __shared__ unsigned short bufA[MT * KS1N];   // 16896 B
  __shared__ unsigned short bufB[MT * KS2];    //  8704 B
  __shared__ float          bufF[MT * FS];     // 16896 B

  const int tid  = threadIdx.x;
  const int lane = tid & 31;
  const int wid  = tid >> 5;
  const int rt   = wid >> 1;
  const int colb = (wid & 1) * 64;
  const long n0  = (long)blockIdx.x * MT;

  // Stage n_in = cat(x, agg): 32 x 256, float4-vectorized.
  for (int i = 0; i < (MT * 64) / BLK; ++i) {   // 64 float4 per row; 16 iters
    int f  = tid + BLK * i;
    int r  = f >> 6;
    int c4 = f & 63;
    int c  = c4 * 4;
    long row = n0 + r;
    float4 v = make_float4(0.f, 0.f, 0.f, 0.f);
    if (row < Ncnt)
      v = (c < 128) ? *(const float4*)(x + row * 128 + c)
                    : *(const float4*)(agg + row * 128 + (c - 128));
    unsigned* dst = (unsigned*)(bufA + r * KS1N + c);
    dst[0] = pack2bf(v.x, v.y);
    dst[1] = pack2bf(v.z, v.w);
  }
  __syncthreads();

  const f32x8 zero = {0.f,0.f,0.f,0.f,0.f,0.f,0.f,0.f};
  f32x8 acc[4];

  acc[0]=zero; acc[1]=zero; acc[2]=zero; acc[3]=zero;
  wave_gemm<256>(wt1, bufA, KS1N, rt, colb, lane, acc);
  store_relu_bf16(bufB, KS2, b1, rt, colb, lane, acc);
  __syncthreads();

  acc[0]=zero; acc[1]=zero; acc[2]=zero; acc[3]=zero;
  wave_gemm<128>(wt2, bufB, KS2, rt, colb, lane, acc);
  store_relu_bf16(bufA, KS2, b2, rt, colb, lane, acc);
  __syncthreads();

  acc[0]=zero; acc[1]=zero; acc[2]=zero; acc[3]=zero;
  wave_gemm<128>(wt3, bufA, KS2, rt, colb, lane, acc);
  store_bias_f32(bufF, FS, b3, rt, colb, lane, acc);
  __syncthreads();

  const int r = tid >> 2;
  const int q = tid & 3;
  long row = n0 + r;
  const float4* rowF4 = (const float4*)(bufF + r * FS + q * 32);
  float s = 0.f, ss = 0.f;
#pragma unroll
  for (int i = 0; i < 8; ++i) {
    float4 h = rowF4[i];
    s  += h.x + h.y + h.z + h.w;
    ss += h.x * h.x + h.y * h.y + h.z * h.z + h.w * h.w;
  }
  s  += __shfl_xor(s, 1, 4);   s  += __shfl_xor(s, 2, 4);
  ss += __shfl_xor(ss, 1, 4);  ss += __shfl_xor(ss, 2, 4);
  float mu  = s * (1.f / 128.f);
  float var = ss * (1.f / 128.f) - mu * mu;
  float rs  = rsqrtf(var + 1e-5f);
  if (row < Ncnt) {
    float4*       xRow4 = (float4*)(x + row * 128 + q * 32);
    const float4* g4    = (const float4*)(gw + q * 32);
    const float4* bt4   = (const float4*)(bt + q * 32);
#pragma unroll
    for (int i = 0; i < 8; ++i) {
      float4 h  = rowF4[i];
      float4 gv = g4[i];
      float4 bv = bt4[i];
      float4 xv = xRow4[i];
      float4 o;
      o.x = xv.x + (h.x - mu) * rs * gv.x + bv.x;
      o.y = xv.y + (h.y - mu) * rs * gv.y + bv.y;
      o.z = xv.z + (h.z - mu) * rs * gv.z + bv.z;
      o.w = xv.w + (h.w - mu) * rs * gv.w + bv.w;
      xRow4[i] = o;
    }
  }
}

// ---------------------------------------------------------------------------
// Helpers: zero a buffer; convert weights f32 [L][K][128] -> bf16 [L][128][K]
// ---------------------------------------------------------------------------
__global__ void zero_kernel(float* __restrict__ p, long n) {
  long i = (long)blockIdx.x * blockDim.x + threadIdx.x;
  if (i < n) p[i] = 0.f;
}

__global__ void cvt_wt_kernel(const float* __restrict__ W,
                              unsigned short* __restrict__ Wt,
                              int Kdim, long total) {
  long i = (long)blockIdx.x * blockDim.x + threadIdx.x;
  if (i >= total) return;
  long per = (long)Kdim * 128;
  int  l   = (int)(i / per);
  long rem = i - (long)l * per;
  int  k   = (int)(rem >> 7);       // /128
  int  n   = (int)(rem & 127);
  Wt[((long)l * 128 + n) * Kdim + k] = f2bf(W[i]);
}

// ---------------------------------------------------------------------------
// Host launcher
// ---------------------------------------------------------------------------
extern "C" void kernel_launch(void* const* d_in, const int* in_sizes, int n_in,
                              void* d_out, int out_size, void* d_ws, size_t ws_size,
                              hipStream_t stream) {
  (void)in_sizes; (void)n_in; (void)out_size; (void)ws_size;
  const int N = 50000, E = 800000, D = 128, L = 4;

  const float* x0   = (const float*)d_in[0];
  const int*   eidx = (const int*)d_in[1];
  const float* ea0  = (const float*)d_in[2];
  const float* eW1  = (const float*)d_in[3];  const float* eb1 = (const float*)d_in[4];
  const float* eW2  = (const float*)d_in[5];  const float* eb2 = (const float*)d_in[6];
  const float* eW3  = (const float*)d_in[7];  const float* eb3 = (const float*)d_in[8];
  const float* eg   = (const float*)d_in[9];  const float* ebt = (const float*)d_in[10];
  const float* nW1  = (const float*)d_in[11]; const float* nb1 = (const float*)d_in[12];
  const float* nW2  = (const float*)d_in[13]; const float* nb2 = (const float*)d_in[14];
  const float* nW3  = (const float*)d_in[15]; const float* nb3 = (const float*)d_in[16];
  const float* ng   = (const float*)d_in[17]; const float* nbt = (const float*)d_in[18];

  float* xbuf  = (float*)d_out;                      // [N,128]
  float* eabuf = (float*)d_out + (size_t)N * D;      // [E,128]

  char* ws = (char*)d_ws;
  auto alignup = [](size_t v) { return (v + 255) & ~(size_t)255; };
  size_t off = 0;
  float* agg = (float*)(ws + off);            off = alignup(off + (size_t)N * D * 4);
  unsigned short* ewt1 = (unsigned short*)(ws + off); off = alignup(off + (size_t)L * 384 * 128 * 2);
  unsigned short* ewt2 = (unsigned short*)(ws + off); off = alignup(off + (size_t)L * 128 * 128 * 2);
  unsigned short* ewt3 = (unsigned short*)(ws + off); off = alignup(off + (size_t)L * 128 * 128 * 2);
  unsigned short* nwt1 = (unsigned short*)(ws + off); off = alignup(off + (size_t)L * 256 * 128 * 2);
  unsigned short* nwt2 = (unsigned short*)(ws + off); off = alignup(off + (size_t)L * 128 * 128 * 2);
  unsigned short* nwt3 = (unsigned short*)(ws + off); off = alignup(off + (size_t)L * 128 * 128 * 2);

  // Working state lives in d_out (matches required output layout).
  hipMemcpyAsync(xbuf,  x0,  (size_t)N * D * 4, hipMemcpyDeviceToDevice, stream);
  hipMemcpyAsync(eabuf, ea0, (size_t)E * D * 4, hipMemcpyDeviceToDevice, stream);

  // Pre-convert weights to transposed bf16 (one-time per launch; tiny).
  {
    long t;
    t = (long)L * 384 * 128; cvt_wt_kernel<<<(int)((t + 255) / 256), 256, 0, stream>>>(eW1, ewt1, 384, t);
    t = (long)L * 128 * 128; cvt_wt_kernel<<<(int)((t + 255) / 256), 256, 0, stream>>>(eW2, ewt2, 128, t);
    t = (long)L * 128 * 128; cvt_wt_kernel<<<(int)((t + 255) / 256), 256, 0, stream>>>(eW3, ewt3, 128, t);
    t = (long)L * 256 * 128; cvt_wt_kernel<<<(int)((t + 255) / 256), 256, 0, stream>>>(nW1, nwt1, 256, t);
    t = (long)L * 128 * 128; cvt_wt_kernel<<<(int)((t + 255) / 256), 256, 0, stream>>>(nW2, nwt2, 128, t);
    t = (long)L * 128 * 128; cvt_wt_kernel<<<(int)((t + 255) / 256), 256, 0, stream>>>(nW3, nwt3, 128, t);
  }

  for (int l = 0; l < L; ++l) {
    long aggN = (long)N * D;
    zero_kernel<<<(int)((aggN + 255) / 256), 256, 0, stream>>>(agg, aggN);

    edge_kernel<<<(E + MT - 1) / MT, BLK, 0, stream>>>(
        xbuf, eidx, eabuf,
        ewt1 + (size_t)l * 384 * 128,
        ewt2 + (size_t)l * 128 * 128,
        ewt3 + (size_t)l * 128 * 128,
        eb1 + l * D, eb2 + l * D, eb3 + l * D, eg + l * D, ebt + l * D,
        agg, E);

    node_kernel<<<(N + MT - 1) / MT, BLK, 0, stream>>>(
        xbuf, agg,
        nwt1 + (size_t)l * 256 * 128,
        nwt2 + (size_t)l * 128 * 128,
        nwt3 + (size_t)l * 128 * 128,
        nb1 + l * D, nb2 + l * D, nb3 + l * D, ng + l * D, nbt + l * D,
        N);
  }
}
